// DynamicFeaturePyramid_3719441678517
// MI455X (gfx1250) — matrix operand order repaired and checked by reading, not compiled, not run on hardware
//
#include <hip/hip_runtime.h>
#include <hip/hip_bf16.h>

typedef __attribute__((ext_vector_type(16))) __bf16 v16bf;
typedef __attribute__((ext_vector_type(8)))  __bf16 v8bf;
typedef __attribute__((ext_vector_type(8)))  float  v8f;

#define WMMA_BF16(A, B, C) \
    __builtin_amdgcn_wmma_f32_16x16x32_bf16(false, (A), false, (B), (short)0, (C), false, false)

// Load a 16-element bf16 fragment as two contiguous 8-element (16B) vectors.
__device__ inline v16bf ld_pair(const __bf16* p0, const __bf16* p1) {
    v8bf a = *(const v8bf*)p0;
    v8bf b = *(const v8bf*)p1;
    v16bf r;
#pragma unroll
    for (int e = 0; e < 8; ++e) { r[e] = a[e]; r[8 + e] = b[e]; }
    return r;
}

// ---------------------------------------------------------------------------
// Kernel 1: build concatenated pyramid feature gf (bf16, pixel-major [B][4096][128])
// channels 0-31 = x1, 32-63 = up(x2,32), 64-95 = up(x3,16), 96-127 = up(x4,8)
// align-corners bilinear upsample to 64x64.
// ---------------------------------------------------------------------------
__global__ __launch_bounds__(256) void gf_build_kernel(
    const float* __restrict__ x1, const float* __restrict__ x2,
    const float* __restrict__ x3, const float* __restrict__ x4,
    __bf16* __restrict__ gf) {
    size_t idx = (size_t)blockIdx.x * 256 + threadIdx.x;          // [0, 8*4096*128)
    int ch  = (int)(idx & 127);
    int pix = (int)((idx >> 7) & 4095);
    int b   = (int)(idx >> 19);
    int h = pix >> 6, w = pix & 63;
    float val;
    if (ch < 32) {
        val = x1[(((size_t)b * 32 + ch) * 64 + h) * 64 + w];
    } else {
        int c = ch & 31;
        int S; const float* xs;
        if (ch < 64)      { S = 32; xs = x2; }
        else if (ch < 96) { S = 16; xs = x3; }
        else              { S = 8;  xs = x4; }
        float step = (float)(S - 1) / 63.0f;
        float fy = h * step, fx = w * step;
        int y0 = (int)floorf(fy); y0 = min(max(y0, 0), S - 1);
        int x0 = (int)floorf(fx); x0 = min(max(x0, 0), S - 1);
        int y1 = min(y0 + 1, S - 1), xx1 = min(x0 + 1, S - 1);
        float wy = fy - (float)y0, wx = fx - (float)x0;
        const float* pl = xs + ((size_t)b * 32 + c) * S * S;
        float top = pl[y0 * S + x0] * (1.f - wy) + pl[y1 * S + x0] * wy;
        float bot = pl[y0 * S + xx1] * (1.f - wy) + pl[y1 * S + xx1] * wy;
        val = top * (1.f - wx) + bot * wx;
    }
    gf[idx] = (__bf16)val;
}

// ---------------------------------------------------------------------------
// Kernel 2: convert q/k/v weights (64x128 f32, [co][c]) to bf16 transposed [c][co]
// ---------------------------------------------------------------------------
__global__ __launch_bounds__(256) void wprep_kernel(
    const float* __restrict__ qw, const float* __restrict__ kw, const float* __restrict__ vw,
    __bf16* __restrict__ wqT, __bf16* __restrict__ wkT, __bf16* __restrict__ wvT) {
    int idx = blockIdx.x * 256 + threadIdx.x;
    if (idx >= 3 * 64 * 128) return;
    int m = idx / 8192, t = idx % 8192;
    int co = t / 128, c = t % 128;
    const float* w = (m == 0) ? qw : (m == 1) ? kw : vw;
    __bf16* o = (m == 0) ? wqT : (m == 1) ? wkT : wvT;
    o[c * 64 + co] = (__bf16)w[co * 128 + c];
}

// ---------------------------------------------------------------------------
// Kernel 3: QKV projection GEMM via v_wmma_f32_16x16x32_bf16.
// Each wave: 16 pixels x 64 out-channels x {q,k,v}, K=128 in 4 chunks of 32.
// q,v stored pixel-major bf16 [B][4096][64]; k stored channel-major [B][64][4096].
// ---------------------------------------------------------------------------
__global__ __launch_bounds__(128) void qkv_kernel(
    const __bf16* __restrict__ gf,
    const __bf16* __restrict__ wqT, const __bf16* __restrict__ wkT, const __bf16* __restrict__ wvT,
    const float* __restrict__ qbias, const float* __restrict__ kbias, const float* __restrict__ vbias,
    __bf16* __restrict__ q_pm, __bf16* __restrict__ k_cm, __bf16* __restrict__ v_pm) {
    int b = blockIdx.x >> 6;
    int pblk = blockIdx.x & 63;
    int wave = threadIdx.x >> 5;
    int lane = threadIdx.x & 31;
    int hl = lane >> 4, r = lane & 15;
    int pbase = pblk * 64 + wave * 16;
    const __bf16* g = gf + ((size_t)b * 4096 + pbase) * 128;

    // A fragments: 16 pixels x 128 channels -> 4 frags of 16x32
    v16bf a[4];
#pragma unroll
    for (int ch = 0; ch < 4; ++ch) {
        int cb = ch * 32;
        a[ch] = ld_pair(g + r * 128 + cb + 8 * hl, g + r * 128 + cb + 16 + 8 * hl);
    }

    const __bf16* Ws[3] = {wqT, wkT, wvT};
    const float* Bi[3]  = {qbias, kbias, vbias};
#pragma unroll
    for (int m = 0; m < 3; ++m) {
#pragma unroll
        for (int co = 0; co < 4; ++co) {
            v8f acc = {};
#pragma unroll
            for (int ch = 0; ch < 4; ++ch) {
                const __bf16* wp = Ws[m] + (size_t)(ch * 32 + 8 * hl) * 64 + co * 16 + r;
                v16bf bw;
#pragma unroll
                for (int e = 0; e < 8; ++e) { bw[e] = wp[e * 64]; bw[8 + e] = wp[(16 + e) * 64]; }
                acc = WMMA_BF16(a[ch], bw, acc);
            }
            float bias = Bi[m][co * 16 + r];
#pragma unroll
            for (int gg = 0; gg < 8; ++gg) {
                int row = pbase + gg + 8 * hl;
                int col = co * 16 + r;
                float v = acc[gg] + bias;
                if (m == 0)      q_pm[((size_t)b * 4096 + row) * 64 + col] = (__bf16)v;
                else if (m == 1) k_cm[((size_t)b * 64 + col) * 4096 + row] = (__bf16)v;
                else             v_pm[((size_t)b * 4096 + row) * 64 + col] = (__bf16)v;
            }
        }
    }
}

// ---------------------------------------------------------------------------
// Kernel 4: flash attention, seq=4096, dim=64, no scale (matches reference).
// Wave owns 16 query rows; loops over key blocks of 32.
// Per block: 4 QK wmmas + 4 PV wmmas. P goes through wave-private LDS tile
// for the C-layout -> A-layout shuffle (same-wave LDS ops are in-order).
// ---------------------------------------------------------------------------
__global__ __launch_bounds__(128) void flash_attn_kernel(
    const __bf16* __restrict__ q, const __bf16* __restrict__ k,
    const __bf16* __restrict__ v, float* __restrict__ O) {
    int b = blockIdx.x >> 6;
    int qblk = blockIdx.x & 63;
    int wave = threadIdx.x >> 5;
    int lane = threadIdx.x & 31;
    int hl = lane >> 4, r = lane & 15;
    int qbase = qblk * 64 + wave * 16;

    const __bf16* qp = q + ((size_t)b * 4096 + qbase) * 64;
    const __bf16* kp = k + (size_t)b * 64 * 4096;
    const __bf16* vp = v + (size_t)b * 4096 * 64;

    __shared__ __align__(16) __bf16 ldsP[4][16 * 32];
    __bf16* P = ldsP[wave];

    // Q A-fragments (16 rows x dims 0..31 / 32..63)
    v16bf aq0 = ld_pair(qp + r * 64 + 8 * hl,      qp + r * 64 + 16 + 8 * hl);
    v16bf aq1 = ld_pair(qp + r * 64 + 32 + 8 * hl, qp + r * 64 + 48 + 8 * hl);

    v8f o0 = {}, o1 = {}, o2 = {}, o3 = {};
    float rm[8], rs[8];
#pragma unroll
    for (int gg = 0; gg < 8; ++gg) { rm[gg] = -3.0e38f; rs[gg] = 0.f; }

    for (int nb = 0; nb < 4096; nb += 32) {
        // K^T B-fragments: two key tiles x two dim chunks
        const __bf16* k0 = kp + (size_t)(8 * hl) * 4096 + nb + r;
        const __bf16* k1 = kp + (size_t)(32 + 8 * hl) * 4096 + nb + r;
        v16bf bk00, bk01, bk10, bk11;
#pragma unroll
        for (int e = 0; e < 8; ++e) {
            bk00[e] = k0[e * 4096];      bk00[8 + e] = k0[(16 + e) * 4096];
            bk01[e] = k1[e * 4096];      bk01[8 + e] = k1[(16 + e) * 4096];
            bk10[e] = k0[e * 4096 + 16]; bk10[8 + e] = k0[(16 + e) * 4096 + 16];
            bk11[e] = k1[e * 4096 + 16]; bk11[8 + e] = k1[(16 + e) * 4096 + 16];
        }

        v8f s0 = {}, s1 = {};
        s0 = WMMA_BF16(aq0, bk00, s0);
        s0 = WMMA_BF16(aq1, bk01, s0);
        s1 = WMMA_BF16(aq0, bk10, s1);
        s1 = WMMA_BF16(aq1, bk11, s1);

        // online softmax update, rows split across wave32 halves per C-layout
        float scale[8];
#pragma unroll
        for (int gg = 0; gg < 8; ++gg) {
            float tm = fmaxf(s0[gg], s1[gg]);
            tm = fmaxf(tm, __shfl_xor(tm, 1, 32));
            tm = fmaxf(tm, __shfl_xor(tm, 2, 32));
            tm = fmaxf(tm, __shfl_xor(tm, 4, 32));
            tm = fmaxf(tm, __shfl_xor(tm, 8, 32));
            float mn = fmaxf(rm[gg], tm);
            float p0 = __expf(s0[gg] - mn);
            float p1 = __expf(s1[gg] - mn);
            float ts = p0 + p1;
            ts += __shfl_xor(ts, 1, 32);
            ts += __shfl_xor(ts, 2, 32);
            ts += __shfl_xor(ts, 4, 32);
            ts += __shfl_xor(ts, 8, 32);
            float sc = __expf(rm[gg] - mn);
            rs[gg] = rs[gg] * sc + ts;
            rm[gg] = mn;
            scale[gg] = sc;
            int row = gg + 8 * hl;
            P[row * 32 + r]      = (__bf16)p0;   // key tile 0
            P[row * 32 + 16 + r] = (__bf16)p1;   // key tile 1
        }
#pragma unroll
        for (int gg = 0; gg < 8; ++gg) {
            o0[gg] *= scale[gg]; o1[gg] *= scale[gg];
            o2[gg] *= scale[gg]; o3[gg] *= scale[gg];
        }

        // reload P as A-fragment (16 queries x 32 keys)
        v16bf ap = ld_pair(P + r * 32 + 8 * hl, P + r * 32 + 16 + 8 * hl);

        // V B-fragments: 32 keys x 4 dim tiles of 16
        const __bf16* vrow = vp + (size_t)(nb + 8 * hl) * 64 + r;
        v16bf bv0, bv1, bv2, bv3;
#pragma unroll
        for (int e = 0; e < 8; ++e) {
            bv0[e] = vrow[e * 64 +  0]; bv0[8 + e] = vrow[(16 + e) * 64 +  0];
            bv1[e] = vrow[e * 64 + 16]; bv1[8 + e] = vrow[(16 + e) * 64 + 16];
            bv2[e] = vrow[e * 64 + 32]; bv2[8 + e] = vrow[(16 + e) * 64 + 32];
            bv3[e] = vrow[e * 64 + 48]; bv3[8 + e] = vrow[(16 + e) * 64 + 48];
        }

        o0 = WMMA_BF16(ap, bv0, o0);
        o1 = WMMA_BF16(ap, bv1, o1);
        o2 = WMMA_BF16(ap, bv2, o2);
        o3 = WMMA_BF16(ap, bv3, o3);
    }

    // epilogue: divide by softmax denominator, store f32 pixel-major [B][4096][64]
#pragma unroll
    for (int gg = 0; gg < 8; ++gg) {
        float inv = 1.0f / rs[gg];
        size_t base = ((size_t)b * 4096 + qbase + gg + 8 * hl) * 64 + r;
        O[base +  0] = o0[gg] * inv;
        O[base + 16] = o1[gg] * inv;
        O[base + 32] = o2[gg] * inv;
        O[base + 48] = o3[gg] * inv;
    }
}

// ---------------------------------------------------------------------------
// Kernel 5: fused out-projection (64->128) + gate MLP (128->32 relu, 32->128 sigmoid).
// One thread per pixel. sw stored channel-major f32 [B][128][4096].
// ---------------------------------------------------------------------------
__global__ __launch_bounds__(128) void post_kernel(
    const float* __restrict__ O,
    const float* __restrict__ ow, const float* __restrict__ obias,
    const float* __restrict__ g1w, const float* __restrict__ g1b,
    const float* __restrict__ g2w, const float* __restrict__ g2b,
    float* __restrict__ sw) {
    int p = blockIdx.x * 128 + threadIdx.x;
    if (p >= 8 * 4096) return;
    int b = p >> 12, pix = p & 4095;

    float o[64];
#pragma unroll
    for (int i = 0; i < 64; ++i) o[i] = O[(size_t)p * 64 + i];

    float hacc[32];
#pragma unroll
    for (int c = 0; c < 32; ++c) hacc[c] = g1b[c];

    for (int j = 0; j < 128; ++j) {
        float gfo = obias[j];
#pragma unroll
        for (int i = 0; i < 64; ++i) gfo += ow[j * 64 + i] * o[i];
#pragma unroll
        for (int c = 0; c < 32; ++c) hacc[c] += g1w[c * 128 + j] * gfo;
    }
#pragma unroll
    for (int c = 0; c < 32; ++c) hacc[c] = fmaxf(hacc[c], 0.f);

    for (int co = 0; co < 128; ++co) {
        float s = g2b[co];
#pragma unroll
        for (int c = 0; c < 32; ++c) s += g2w[co * 32 + c] * hacc[c];
        sw[((size_t)b * 128 + co) * 4096 + pix] = 1.f / (1.f + __expf(-s));
    }
}

// ---------------------------------------------------------------------------
// Kernel 6: gate application. Gate plane (64x64) sampled with align-corners
// bilinear at the level's resolution (identity for S=64).
// ---------------------------------------------------------------------------
__global__ __launch_bounds__(256) void gate_kernel(
    const float* __restrict__ x, const float* __restrict__ sw,
    float* __restrict__ out, int S, int chbase, int N) {
    int idx = blockIdx.x * 256 + threadIdx.x;
    if (idx >= N) return;
    int w = idx % S;
    int h = (idx / S) % S;
    int c = (idx / (S * S)) % 32;
    int b = idx / (32 * S * S);
    float step = 63.0f / (float)(S - 1);
    float fy = h * step, fx = w * step;
    int y0 = (int)floorf(fy); y0 = min(max(y0, 0), 63);
    int x0 = (int)floorf(fx); x0 = min(max(x0, 0), 63);
    int y1 = min(y0 + 1, 63), xx1 = min(x0 + 1, 63);
    float wy = fy - (float)y0, wx = fx - (float)x0;
    const float* pl = sw + ((size_t)b * 128 + chbase + c) * 4096;
    float top = pl[y0 * 64 + x0] * (1.f - wy) + pl[y1 * 64 + x0] * wy;
    float bot = pl[y0 * 64 + xx1] * (1.f - wy) + pl[y1 * 64 + xx1] * wy;
    float g = top * (1.f - wx) + bot * wx;
    out[idx] = x[idx] * (1.f + g);
}

// ---------------------------------------------------------------------------
extern "C" void kernel_launch(void* const* d_in, const int* in_sizes, int n_in,
                              void* d_out, int out_size, void* d_ws, size_t ws_size,
                              hipStream_t stream) {
    const float* x1 = (const float*)d_in[0];
    const float* x2 = (const float*)d_in[1];
    const float* x3 = (const float*)d_in[2];
    const float* x4 = (const float*)d_in[3];
    const float* qw = (const float*)d_in[4];
    const float* qb = (const float*)d_in[5];
    const float* kw = (const float*)d_in[6];
    const float* kb = (const float*)d_in[7];
    const float* vw = (const float*)d_in[8];
    const float* vb = (const float*)d_in[9];
    const float* ow = (const float*)d_in[10];
    const float* ob = (const float*)d_in[11];
    const float* g1w = (const float*)d_in[12];
    const float* g1b = (const float*)d_in[13];
    const float* g2w = (const float*)d_in[14];
    const float* g2b = (const float*)d_in[15];
    float* out = (float*)d_out;

    char* ws = (char*)d_ws;
    __bf16* gf   = (__bf16*)(ws);                         // 8*4096*128*2 = 8 MB
    __bf16* q_pm = (__bf16*)(ws + 8388608);               // 4 MB
    __bf16* k_cm = (__bf16*)(ws + 12582912);              // 4 MB
    __bf16* v_pm = (__bf16*)(ws + 16777216);              // 4 MB
    float*  Obuf = (float*) (ws + 20971520);              // 8 MB
    float*  swb  = (float*) (ws + 29360128);              // 16 MB
    __bf16* wqT  = (__bf16*)(ws + 46137344);              // 16 KB
    __bf16* wkT  = (__bf16*)(ws + 46153728);              // 16 KB
    __bf16* wvT  = (__bf16*)(ws + 46170112);              // 16 KB

    // 1. build concat feature (bf16 pixel-major)
    gf_build_kernel<<<(8 * 4096 * 128) / 256, 256, 0, stream>>>(x1, x2, x3, x4, gf);
    // 2. weight prep
    wprep_kernel<<<(3 * 64 * 128 + 255) / 256, 256, 0, stream>>>(qw, kw, vw, wqT, wkT, wvT);
    // 3. QKV projections (WMMA bf16)
    qkv_kernel<<<8 * 64, 128, 0, stream>>>(gf, wqT, wkT, wvT, qb, kb, vb, q_pm, k_cm, v_pm);
    // 4. flash attention (WMMA bf16)
    flash_attn_kernel<<<8 * 64, 128, 0, stream>>>(q_pm, k_cm, v_pm, Obuf);
    // 5. out-projection + gate MLP
    post_kernel<<<(8 * 4096) / 128, 128, 0, stream>>>(Obuf, ow, ob, g1w, g1b, g2w, g2b, swb);
    // 6. apply gates; outputs concatenated flat in return order
    int N1 = 8 * 32 * 64 * 64;   // 1048576
    int N2 = 8 * 32 * 32 * 32;   // 262144
    int N3 = 8 * 32 * 16 * 16;   // 65536
    int N4 = 8 * 32 * 8 * 8;     // 16384
    gate_kernel<<<(N1 + 255) / 256, 256, 0, stream>>>(x1, swb, out, 64, 0, N1);
    gate_kernel<<<(N2 + 255) / 256, 256, 0, stream>>>(x2, swb, out + N1, 32, 32, N2);
    gate_kernel<<<(N3 + 255) / 256, 256, 0, stream>>>(x3, swb, out + N1 + N2, 16, 64, N3);
    gate_kernel<<<(N4 + 255) / 256, 256, 0, stream>>>(x4, swb, out + N1 + N2 + N3, 8, 96, N4);
}
